// TransformerEncoderLayer_76347338653913
// MI455X (gfx1250) — matrix operand order repaired and checked by reading, not compiled, not run on hardware
//
#include <hip/hip_runtime.h>
#include <hip/hip_bf16.h>

// ---------------------------------------------------------------------------
// Types for CDNA5 WMMA (wave32, v_wmma_f32_16x16x32_f16)
// ---------------------------------------------------------------------------
typedef _Float16 h16;
typedef __attribute__((ext_vector_type(16))) _Float16    v16h;
typedef __attribute__((ext_vector_type(8)))  float       v8f;
typedef __attribute__((ext_vector_type(4)))  unsigned int u32x4;
typedef __attribute__((ext_vector_type(4)))  float       f32x4;

union Frag {
    v16h     v;
    _Float16 h[16];
    u32x4    q[2];
};

__device__ __forceinline__ v8f wmma_f16(v16h a, v16h b, v8f c) {
    return __builtin_amdgcn_wmma_f32_16x16x32_f16(false, a, false, b,
                                                  (short)0, c, false, false);
}

// Problem constants
constexpr int cB  = 128;
constexpr int cN  = 256;
constexpr int cC  = 768;
constexpr int cH  = 12;
constexpr int cHD = 64;
constexpr int cFF = 3072;
constexpr float cEPS   = 1e-5f;
constexpr float cSCALE = 0.125f;   // HD^-0.5

// ---------------------------------------------------------------------------
// f32 -> f16 convert (weights)
// ---------------------------------------------------------------------------
__global__ __launch_bounds__(256) void cvt_f16(const float* __restrict__ in,
                                               h16* __restrict__ out, long long n) {
    long long i = (long long)blockIdx.x * blockDim.x + threadIdx.x;
    long long stride = (long long)gridDim.x * blockDim.x;
    for (; i < n; i += stride) out[i] = (h16)in[i];
}

// ---------------------------------------------------------------------------
// LayerNorm over C=768, one block per row; writes f16 (+optional f32)
// ---------------------------------------------------------------------------
__global__ __launch_bounds__(256) void layernorm_k(const float* __restrict__ in,
                                                   const float* __restrict__ g,
                                                   const float* __restrict__ bt,
                                                   float* __restrict__ outF,
                                                   h16* __restrict__ outH) {
    long long row = blockIdx.x;
    const float* x = in + row * cC;
    int tid = threadIdx.x;
    float v0 = x[tid], v1 = x[tid + 256], v2 = x[tid + 512];
    __shared__ float red[256];
    red[tid] = v0 + v1 + v2;
    __syncthreads();
    for (int off = 128; off > 0; off >>= 1) {
        if (tid < off) red[tid] += red[tid + off];
        __syncthreads();
    }
    float mu = red[0] * (1.0f / cC);
    __syncthreads();
    float d0 = v0 - mu, d1 = v1 - mu, d2 = v2 - mu;
    red[tid] = d0 * d0 + d1 * d1 + d2 * d2;
    __syncthreads();
    for (int off = 128; off > 0; off >>= 1) {
        if (tid < off) red[tid] += red[tid + off];
        __syncthreads();
    }
    float rstd = rsqrtf(red[0] * (1.0f / cC) + cEPS);
    float y0 = d0 * rstd * g[tid]       + bt[tid];
    float y1 = d1 * rstd * g[tid + 256] + bt[tid + 256];
    float y2 = d2 * rstd * g[tid + 512] + bt[tid + 512];
    outH[row * cC + tid]       = (h16)y0;
    outH[row * cC + tid + 256] = (h16)y1;
    outH[row * cC + tid + 512] = (h16)y2;
    if (outF) {
        outF[row * cC + tid]       = y0;
        outF[row * cC + tid + 256] = y1;
        outF[row * cC + tid + 512] = y2;
    }
}

// ---------------------------------------------------------------------------
// k-prep: k_ext f32[256][64] -> f16 copy, ksum[64], Gram G = k^T k [64][64]
// ---------------------------------------------------------------------------
__global__ __launch_bounds__(256) void kprep_k(const float* __restrict__ k,
                                               h16* __restrict__ kh,
                                               float* __restrict__ ksum,
                                               float* __restrict__ G) {
    int tid = threadIdx.x;
    for (int i = tid; i < cN * cHD; i += 256) kh[i] = (h16)k[i];
    if (tid < 64) {
        float s = 0.f;
        for (int m = 0; m < cN; ++m) s += k[m * cHD + tid];
        ksum[tid] = s;
    }
    for (int t = 0; t < 16; ++t) {
        int gidx = tid * 16 + t;
        int d = gidx >> 6, e = gidx & 63;
        float s = 0.f;
        for (int m = 0; m < cN; ++m) s += k[m * cHD + d] * k[m * cHD + e];
        G[gidx] = s;
    }
}

__global__ void zero_stats_k(float* stats) {
    if (threadIdx.x < 48) stats[threadIdx.x] = 0.f;
}

// ---------------------------------------------------------------------------
// BN statistics: per head h, over all (b,n) rows accumulate
//   sum_m s = q.ksum          sum_m s^2 = q^T G q
//   sum_d v,  sum_d v^2
// grid (128, 12), thread = one (b,n) row.
// ---------------------------------------------------------------------------
__global__ __launch_bounds__(256) void attn_stats_k(const float* __restrict__ qv,
                                                    const float* __restrict__ G,
                                                    const float* __restrict__ ksum,
                                                    float* __restrict__ stats) {
    int hid = blockIdx.y;
    int tid = threadIdx.x;
    long long rowIdx = (long long)blockIdx.x * 256 + tid;   // b*256+n
    __shared__ float Gs[64 * 64];
    __shared__ float ks[64];
    for (int i = tid; i < 4096; i += 256) Gs[i] = G[i];
    if (tid < 64) ks[tid] = ksum[tid];
    __syncthreads();

    const float* qrow = qv + rowIdx * (2 * cC) + hid * cHD;
    const float* vrow = qv + rowIdx * (2 * cC) + cC + hid * cHD;
    float s1 = 0.f, vs1 = 0.f, vs2 = 0.f;
    for (int d = 0; d < 64; ++d) {
        s1 += qrow[d] * ks[d];
        float vv = vrow[d];
        vs1 += vv;
        vs2 += vv * vv;
    }
    float qgq = 0.f;
    for (int d = 0; d < 64; ++d) {
        float r = 0.f;
        #pragma unroll
        for (int e = 0; e < 64; ++e) r += Gs[d * 64 + e] * qrow[e];
        qgq += qrow[d] * r;
    }
    __shared__ float red[256];
    float vals[4] = {s1, qgq, vs1, vs2};
    for (int i = 0; i < 4; ++i) {
        red[tid] = vals[i];
        __syncthreads();
        for (int off = 128; off > 0; off >>= 1) {
            if (tid < off) red[tid] += red[tid + off];
            __syncthreads();
        }
        if (tid == 0) atomicAdd(&stats[hid * 4 + i], red[0]);
        __syncthreads();
    }
}

__global__ void finalize_stats_k(const float* __restrict__ stats,
                                 const float* __restrict__ bn_g,
                                 const float* __restrict__ bn_b,
                                 float* __restrict__ deriv) {
    int t = threadIdx.x;
    if (t < cH) {
        float cS = (float)cB * cN * cN;
        float mS = stats[t * 4 + 0] / cS;
        float vS = stats[t * 4 + 1] / cS - mS * mS;
        float sS = bn_g[t] * rsqrtf(vS + cEPS);
        deriv[t * 4 + 0] = sS;
        deriv[t * 4 + 1] = bn_b[t] - mS * sS;
        float cV = (float)cB * cN * cHD;
        float mV = stats[t * 4 + 2] / cV;
        float vV = stats[t * 4 + 3] / cV - mV * mV;
        float sV = bn_g[t] * rsqrtf(vV + cEPS);
        deriv[t * 4 + 2] = sV;
        deriv[t * 4 + 3] = bn_b[t] - mV * sV;
    }
}

// ---------------------------------------------------------------------------
// Generic WMMA GEMM:  C[M,Nd] = A[M,K] * W[Nd,K]^T  (both f16 row-major)
// Block tile 128x128, BK=32, 8 waves (4 over M, 2 over N), 2x4 WMMA/wave.
// Epilogue modes:
//   0: store f32 [M][Nd]                                    (qv)
//   1: +bias[col], store f16 transposed p1T[b][c][n]        (proj)
//   2: batched (z=b): +bias[row] + x residual, store f32    (conv)
//   3: +bias[col], exact GELU, store f16 [M][Nd]            (fc1)
//   4: +bias[col] + resid, store f32 -> d_out               (fc2)
// ---------------------------------------------------------------------------
__global__ __launch_bounds__(256) void gemm_f16_k(const h16* __restrict__ A,
                                                  long long strideA,
                                                  const h16* __restrict__ W,
                                                  long long strideW,
                                                  int Kdim, int Nd, int mode,
                                                  const float* __restrict__ bias,
                                                  const float* __restrict__ resid,
                                                  float* __restrict__ outF,
                                                  h16* __restrict__ outH) {
    constexpr int LSTR = 40;                  // padded LDS row stride (halfs)
    __shared__ h16 As[128 * LSTR];
    __shared__ h16 Ws[128 * LSTR];

    int bz = blockIdx.z;
    A += (long long)bz * strideA;
    W += (long long)bz * strideW;

    int tid = threadIdx.x;
    int lane = tid & 31, wave = tid >> 5;
    int waveM = wave & 3, waveN = wave >> 2;
    int lh = lane >> 4, ll = lane & 15;
    int rowBase = blockIdx.y * 128;
    int colBase = blockIdx.x * 128;

    v8f acc[2][4];
    v8f vz = {0.f, 0.f, 0.f, 0.f, 0.f, 0.f, 0.f, 0.f};
    #pragma unroll
    for (int mi = 0; mi < 2; ++mi)
        #pragma unroll
        for (int ni = 0; ni < 4; ++ni) acc[mi][ni] = vz;

    for (int k0 = 0; k0 < Kdim; k0 += 32) {
        // cooperative tile load: 128x32 halfs for A and W (8 halfs/thread x2)
        #pragma unroll
        for (int it = 0; it < 2; ++it) {
            int idx = (tid + it * 256) * 8;
            int r = idx >> 5, kk = idx & 31;
            *(u32x4*)(&As[r * LSTR + kk]) =
                *(const u32x4*)(A + (long long)(rowBase + r) * Kdim + k0 + kk);
            *(u32x4*)(&Ws[r * LSTR + kk]) =
                *(const u32x4*)(W + (long long)(colBase + r) * Kdim + k0 + kk);
        }
        __syncthreads();

        Frag af[2], wf[4];
        #pragma unroll
        for (int mi = 0; mi < 2; ++mi) {      // A: two 8-half spans per lane
            int r = waveM * 32 + mi * 16 + ll;
            af[mi].q[0] = *(const u32x4*)(&As[r * LSTR + lh * 8]);
            af[mi].q[1] = *(const u32x4*)(&As[r * LSTR + 16 + lh * 8]);
        }
        #pragma unroll
        for (int ni = 0; ni < 4; ++ni) {      // B: two 8-half spans per lane
            int r = waveN * 64 + ni * 16 + ll;
            wf[ni].q[0] = *(const u32x4*)(&Ws[r * LSTR + lh * 8]);
            wf[ni].q[1] = *(const u32x4*)(&Ws[r * LSTR + 16 + lh * 8]);
        }
        #pragma unroll
        for (int mi = 0; mi < 2; ++mi)
            #pragma unroll
            for (int ni = 0; ni < 4; ++ni)
                acc[mi][ni] = wmma_f16(af[mi].v, wf[ni].v, acc[mi][ni]);
        __syncthreads();
    }

    // epilogue: C layout => col = ll, row-in-tile = j + 8*lh
    #pragma unroll
    for (int mi = 0; mi < 2; ++mi)
        #pragma unroll
        for (int ni = 0; ni < 4; ++ni)
            #pragma unroll
            for (int j = 0; j < 8; ++j) {
                int row = rowBase + waveM * 32 + mi * 16 + j + 8 * lh;
                int col = colBase + waveN * 64 + ni * 16 + ll;
                float v = acc[mi][ni][j];
                if (mode == 0) {
                    outF[(long long)row * Nd + col] = v;
                } else if (mode == 1) {
                    v += bias[col];
                    int bb = row >> 8, nn = row & 255;
                    outH[((long long)bb * cC + col) * cN + nn] = (h16)v;
                } else if (mode == 2) {
                    long long o = ((long long)bz * cN + row) * cC + col;
                    outF[o] = v + bias[row] + resid[o];
                } else if (mode == 3) {
                    v += bias[col];
                    float gg = 0.5f * v * (1.0f + erff(v * 0.70710678118f));
                    outH[(long long)row * Nd + col] = (h16)gg;
                } else {
                    long long o = (long long)row * cC + col;
                    outF[o] = v + bias[col] + resid[o];
                }
            }
}

// ---------------------------------------------------------------------------
// Fused attention (flash-style), one block = (b, head, half of 128 rows).
// Phase 1: BN-normalize V into LDS transposed [d][m].
// Per wave (16 q-rows): WMMA q*k^T -> BN+bias+scale -> softmax (shfl within
// 16-lane halves) -> stage p through per-wave LDS -> WMMA p*v -> f16 out.
// ---------------------------------------------------------------------------
__global__ __launch_bounds__(256) void attention_k(const float* __restrict__ qv,
                                                   const h16* __restrict__ kh,
                                                   const float* __restrict__ deriv,
                                                   const float* __restrict__ abias,
                                                   h16* __restrict__ outH) {
    constexpr int VSTR = 264;                 // padded: 256 + 8
    __shared__ h16 vt[cHD * VSTR];            // V^T, BN-normalized, f16
    __shared__ h16 plds[8 * 16 * 40];         // per-wave p staging

    int blk = blockIdx.x;
    int half = blk & 1;
    int bh = blk >> 1;
    int hid = bh % cH;
    int b = bh / cH;
    int tid = threadIdx.x, lane = tid & 31, wave = tid >> 5;
    int lh = lane >> 4, ll = lane & 15;

    float sS = deriv[hid * 4 + 0], bS = deriv[hid * 4 + 1];
    float sV = deriv[hid * 4 + 2], bV = deriv[hid * 4 + 3];

    // phase 1: all 256 m-rows of V for this (b,h)
    for (int i = tid; i < cN * cHD; i += 256) {
        int m = i >> 6, d = i & 63;
        float vv = qv[((long long)b * cN + m) * (2 * cC) + cC + hid * cHD + d];
        vt[d * VSTR + m] = (h16)(vv * sV + bV);
    }
    __syncthreads();

    int nrow = half * 128 + wave * 16;

    // q fragments (load f32, convert to f16 in-register)
    Frag qf[2];
    #pragma unroll
    for (int dk = 0; dk < 2; ++dk) {
        const float* qp = qv + ((long long)b * cN + nrow + ll) * (2 * cC)
                          + hid * cHD + dk * 32 + lh * 8;
        f32x4 a0 = *(const f32x4*)(qp);
        f32x4 a1 = *(const f32x4*)(qp + 4);
        f32x4 a2 = *(const f32x4*)(qp + 16);
        f32x4 a3 = *(const f32x4*)(qp + 20);
        #pragma unroll
        for (int j = 0; j < 4; ++j) {
            qf[dk].h[j]      = (h16)a0[j];
            qf[dk].h[4 + j]  = (h16)a1[j];
            qf[dk].h[8 + j]  = (h16)a2[j];
            qf[dk].h[12 + j] = (h16)a3[j];
        }
    }

    // scores: 16 m-tiles x (K=64 as two k-chunks)
    v8f sc[16];
    v8f vz = {0.f, 0.f, 0.f, 0.f, 0.f, 0.f, 0.f, 0.f};
    #pragma unroll
    for (int mt = 0; mt < 16; ++mt) {
        v8f c = vz;
        #pragma unroll
        for (int dk = 0; dk < 2; ++dk) {
            const h16* kp = kh + (mt * 16 + ll) * cHD + dk * 32 + lh * 8;
            Frag kf;
            kf.q[0] = *(const u32x4*)(kp);
            kf.q[1] = *(const u32x4*)(kp + 16);
            c = wmma_f16(qf[dk].v, kf.v, c);
        }
        sc[mt] = c;
    }

    // BN + attn_bias + scale
    #pragma unroll
    for (int mt = 0; mt < 16; ++mt) {
        float ab = abias[mt * 16 + ll];
        #pragma unroll
        for (int j = 0; j < 8; ++j)
            sc[mt][j] = (sc[mt][j] * sS + bS + ab) * cSCALE;
    }

    // softmax over m=256 (row lives in one 16-lane half)
    #pragma unroll
    for (int j = 0; j < 8; ++j) {
        float m = -1e30f;
        #pragma unroll
        for (int mt = 0; mt < 16; ++mt) m = fmaxf(m, sc[mt][j]);
        #pragma unroll
        for (int off = 8; off > 0; off >>= 1) m = fmaxf(m, __shfl_xor(m, off, 32));
        float s = 0.f;
        #pragma unroll
        for (int mt = 0; mt < 16; ++mt) {
            float e = expf(sc[mt][j] - m);
            sc[mt][j] = e;
            s += e;
        }
        #pragma unroll
        for (int off = 8; off > 0; off >>= 1) s += __shfl_xor(s, off, 32);
        float r = 1.0f / s;
        #pragma unroll
        for (int mt = 0; mt < 16; ++mt) sc[mt][j] *= r;
    }

    // attn @ V_bn : loop over m in chunks of 32
    v8f oacc[4];
    #pragma unroll
    for (int dt = 0; dt < 4; ++dt) oacc[dt] = vz;
    h16* pw = plds + wave * 16 * 40;
    for (int mc = 0; mc < 8; ++mc) {
        #pragma unroll
        for (int t = 0; t < 2; ++t) {
            int mt = mc * 2 + t;
            #pragma unroll
            for (int j = 0; j < 8; ++j)
                pw[(j + 8 * lh) * 40 + t * 16 + ll] = (h16)sc[mt][j];
        }
        asm volatile("s_wait_dscnt 0" ::: "memory");  // same-wave DS ordering
        Frag pa;
        pa.q[0] = *(const u32x4*)(&pw[ll * 40 + lh * 8]);
        pa.q[1] = *(const u32x4*)(&pw[ll * 40 + 16 + lh * 8]);
        #pragma unroll
        for (int dt = 0; dt < 4; ++dt) {
            const h16* vp = &vt[(dt * 16 + ll) * VSTR + mc * 32 + lh * 8];
            Frag vf;
            vf.q[0] = *(const u32x4*)(vp);
            vf.q[1] = *(const u32x4*)(vp + 16);
            oacc[dt] = wmma_f16(pa.v, vf.v, oacc[dt]);
        }
    }

    // store o -> [b, n, h*64+d] as f16 for proj GEMM
    #pragma unroll
    for (int dt = 0; dt < 4; ++dt)
        #pragma unroll
        for (int j = 0; j < 8; ++j) {
            int n = nrow + j + 8 * lh;
            int d = dt * 16 + ll;
            outH[((long long)b * cN + n) * cC + hid * cHD + d] = (h16)oacc[dt][j];
        }
}

// ---------------------------------------------------------------------------
// Host: orchestrate the pipeline (all launches on `stream`, ws-allocated)
// ---------------------------------------------------------------------------
extern "C" void kernel_launch(void* const* d_in, const int* in_sizes, int n_in,
                              void* d_out, int out_size, void* d_ws, size_t ws_size,
                              hipStream_t stream) {
    (void)in_sizes; (void)n_in; (void)out_size; (void)ws_size;
    const float* x         = (const float*)d_in[0];
    const float* pre_gamma = (const float*)d_in[1];
    const float* pre_beta  = (const float*)d_in[2];
    const float* qv_w      = (const float*)d_in[3];
    const float* attn_bias = (const float*)d_in[4];
    const float* k_ext     = (const float*)d_in[5];
    const float* bn_gamma  = (const float*)d_in[6];
    const float* bn_beta   = (const float*)d_in[7];
    const float* proj_w    = (const float*)d_in[8];
    const float* proj_b    = (const float*)d_in[9];
    const float* conv_w    = (const float*)d_in[10];
    const float* conv_b    = (const float*)d_in[11];
    const float* norm_gamma = (const float*)d_in[12];
    const float* norm_beta  = (const float*)d_in[13];
    const float* fc1_w     = (const float*)d_in[14];
    const float* fc1_b     = (const float*)d_in[15];
    const float* fc2_w     = (const float*)d_in[16];
    const float* fc2_b     = (const float*)d_in[17];
    float* out = (float*)d_out;

    char* ws = (char*)d_ws;
    size_t off = 0;
    auto alloc = [&](size_t bytes) -> char* {
        char* p = ws + off;
        off += (bytes + 255) & ~(size_t)255;
        return p;
    };
    const long long MT = (long long)cB * cN;               // 32768 token rows
    h16*   qvw_h   = (h16*)alloc((size_t)2 * cC * cC * 2);
    h16*   projw_h = (h16*)alloc((size_t)cC * cC * 2);
    h16*   convw_h = (h16*)alloc((size_t)cN * cN * 2);
    h16*   fc1w_h  = (h16*)alloc((size_t)cFF * cC * 2);
    h16*   fc2w_h  = (h16*)alloc((size_t)cC * cFF * 2);
    h16*   k_h     = (h16*)alloc((size_t)cN * cHD * 2);
    float* ksum_f  = (float*)alloc(64 * 4);
    float* G_f     = (float*)alloc(4096 * 4);
    float* stats_f = (float*)alloc(48 * 4);
    float* deriv_f = (float*)alloc(48 * 4);
    h16*   h_h     = (h16*)alloc((size_t)MT * cC * 2);     // LN1 out; reused: p1T
    float* qv_f    = (float*)alloc((size_t)MT * 2 * cC * 4); // reused: h1 (f16)
    h16*   o_h     = (h16*)alloc((size_t)MT * cC * 2);     // reused: src_ln f16
    float* src_f   = (float*)alloc((size_t)MT * cC * 4);
    float* srcln_f = (float*)alloc((size_t)MT * cC * 4);
    h16*   p1T_h   = h_h;                                  // [b][c][n] f16
    h16*   h1_h    = (h16*)qv_f;                           // fc1 out f16
    h16*   srcln_h = o_h;

    // 1) weight conversions + k-prep + stat zeroing
    cvt_f16<<<2048, 256, 0, stream>>>(qv_w, qvw_h, (long long)2 * cC * cC);
    cvt_f16<<<1024, 256, 0, stream>>>(proj_w, projw_h, (long long)cC * cC);
    cvt_f16<<<256, 256, 0, stream>>>(conv_w, convw_h, (long long)cN * cN);
    cvt_f16<<<4096, 256, 0, stream>>>(fc1_w, fc1w_h, (long long)cFF * cC);
    cvt_f16<<<4096, 256, 0, stream>>>(fc2_w, fc2w_h, (long long)cC * cFF);
    kprep_k<<<1, 256, 0, stream>>>(k_ext, k_h, ksum_f, G_f);
    zero_stats_k<<<1, 64, 0, stream>>>(stats_f);

    // 2) pre-LN -> f16
    layernorm_k<<<(int)MT, 256, 0, stream>>>(x, pre_gamma, pre_beta, nullptr, h_h);

    // 3) qv GEMM: [32768,768] x [1536,768]^T -> f32
    gemm_f16_k<<<dim3(1536 / 128, (int)MT / 128, 1), 256, 0, stream>>>(
        h_h, 0, qvw_h, 0, cC, 2 * cC, 0, nullptr, nullptr, qv_f, nullptr);

    // 4) BN statistics + finalize
    attn_stats_k<<<dim3((int)MT / 256, cH), 256, 0, stream>>>(qv_f, G_f, ksum_f, stats_f);
    finalize_stats_k<<<1, 32, 0, stream>>>(stats_f, bn_gamma, bn_beta, deriv_f);

    // 5) fused attention -> o f16
    attention_k<<<cB * cH * 2, 256, 0, stream>>>(qv_f, k_h, deriv_f, attn_bias, o_h);

    // 6) proj GEMM: +proj_b, store f16 transposed p1T[b][c][n]
    gemm_f16_k<<<dim3(cC / 128, (int)MT / 128, 1), 256, 0, stream>>>(
        o_h, 0, projw_h, 0, cC, cC, 1, proj_b, nullptr, nullptr, p1T_h);

    // 7) conv GEMM (batched over b): conv_w[m][n] x p1T[b][c][n]^T
    gemm_f16_k<<<dim3(cC / 128, cN / 128, cB), 256, 0, stream>>>(
        convw_h, 0, p1T_h, (long long)cC * cN, cN, cC, 2, conv_b, x, src_f, nullptr);

    // 8) mid-LN -> f32 + f16
    layernorm_k<<<(int)MT, 256, 0, stream>>>(src_f, norm_gamma, norm_beta, srcln_f, srcln_h);

    // 9) fc1 GEMM: +fc1_b, exact GELU -> f16
    gemm_f16_k<<<dim3(cFF / 128, (int)MT / 128, 1), 256, 0, stream>>>(
        srcln_h, 0, fc1w_h, 0, cC, cFF, 3, fc1_b, nullptr, nullptr, h1_h);

    // 10) fc2 GEMM: +fc2_b + src_ln residual -> d_out f32
    gemm_f16_k<<<dim3(cC / 128, (int)MT / 128, 1), 256, 0, stream>>>(
        h1_h, 0, fc2w_h, 0, cFF, cC, 4, fc2_b, srcln_f, out, nullptr);
}
